// M_EfficientSelfAtten2_33706903339568
// MI455X (gfx1250) — matrix operand order repaired and checked by reading, not compiled, not run on hardware
//
#include <hip/hip_runtime.h>
#include <hip/hip_bf16.h>
#include <math.h>

typedef __attribute__((ext_vector_type(16))) _Float16 v16h;
typedef __attribute__((ext_vector_type(8)))  float    v8f;
typedef unsigned int   u32;
typedef unsigned short u16;

#define B_  8
#define N_  7680
#define M1_ 960
#define M2_ 2304

union F8  { v8f  v; float f[8];  };
union H16 { v16h v; u32   u[8];  };

static __device__ __forceinline__ u16 f32_to_h16(float x) {
    _Float16 h = (_Float16)x;
    return __builtin_bit_cast(u16, h);
}

// 16-lane-row butterfly reductions via DPP16 (pure VALU, no LDS traffic).
// Masks {1,2,7,15} generate the full 16-lane group with disjoint cosets, so
// both max and sum butterflies are exact.
#define DPP_XOR1  0xB1   // quad_perm [1,0,3,2]
#define DPP_XOR2  0x4E   // quad_perm [2,3,0,1]
#define DPP_XOR7  0x141  // row_half_mirror
#define DPP_XOR15 0x140  // row_mirror

static __device__ __forceinline__ float row_max16(float x) {
    int t;
    t = __builtin_amdgcn_update_dpp(0, __builtin_bit_cast(int, x), DPP_XOR1,  0xF, 0xF, true);
    x = fmaxf(x, __builtin_bit_cast(float, t));
    t = __builtin_amdgcn_update_dpp(0, __builtin_bit_cast(int, x), DPP_XOR2,  0xF, 0xF, true);
    x = fmaxf(x, __builtin_bit_cast(float, t));
    t = __builtin_amdgcn_update_dpp(0, __builtin_bit_cast(int, x), DPP_XOR7,  0xF, 0xF, true);
    x = fmaxf(x, __builtin_bit_cast(float, t));
    t = __builtin_amdgcn_update_dpp(0, __builtin_bit_cast(int, x), DPP_XOR15, 0xF, 0xF, true);
    x = fmaxf(x, __builtin_bit_cast(float, t));
    return x;
}

static __device__ __forceinline__ float row_sum16(float x) {
    int t;
    t = __builtin_amdgcn_update_dpp(0, __builtin_bit_cast(int, x), DPP_XOR1,  0xF, 0xF, true);
    x += __builtin_bit_cast(float, t);
    t = __builtin_amdgcn_update_dpp(0, __builtin_bit_cast(int, x), DPP_XOR2,  0xF, 0xF, true);
    x += __builtin_bit_cast(float, t);
    t = __builtin_amdgcn_update_dpp(0, __builtin_bit_cast(int, x), DPP_XOR7,  0xF, 0xF, true);
    x += __builtin_bit_cast(float, t);
    t = __builtin_amdgcn_update_dpp(0, __builtin_bit_cast(int, x), DPP_XOR15, 0xF, 0xF, true);
    x += __builtin_bit_cast(float, t);
    return x;
}

// ---------------------------------------------------------------------------
// Kernel 1: Q projection.  q = (x @ q_w + q_b) * d^-0.5, split into 2 heads,
// stored f16 row-major [head][B*N][32].
// ---------------------------------------------------------------------------
__global__ void qproj_kernel(const float* __restrict__ x,
                             const float* __restrict__ qw,
                             const float* __restrict__ qb,
                             _Float16* __restrict__ Q) {
    __shared__ float xs[4][64];
    const int slot  = threadIdx.x >> 6;
    const int j     = threadIdx.x & 63;
    const int token = blockIdx.x * 4 + slot;          // flat over B*N
    xs[slot][j] = x[(size_t)token * 64 + j];
    __syncthreads();
    const float* xr = xs[slot];
    float acc = qb[j];
#pragma unroll 8
    for (int c = 0; c < 64; ++c) acc = fmaf(xr[c], qw[c * 64 + j], acc);
    acc *= 0.17677669529663687f;                      // 32^-0.5 folded into Q
    const int head = j >> 5, dd = j & 31;
    Q[((size_t)head * (B_ * N_) + token) * 32 + dd] = (_Float16)acc;
}

// ---------------------------------------------------------------------------
// Kernel 2: build K (row-major f16 [B][M][32]) and V^T (f16 [B][32][M]).
// One block per (batch, token).  Patch-gather into LDS, conv-as-matmul,
// LayerNorm, exact GELU, kv projection.
// ---------------------------------------------------------------------------
__global__ void build_kv_kernel(const float* __restrict__ x,
                                const float* __restrict__ wA, const float* __restrict__ bA,
                                const float* __restrict__ wB, const float* __restrict__ bB,
                                const float* __restrict__ wC, const float* __restrict__ bC,
                                const float* __restrict__ lnw, const float* __restrict__ lnb,
                                const float* __restrict__ kvw, const float* __restrict__ kvb,
                                _Float16* __restrict__ K, _Float16* __restrict__ VT,
                                int M, int branch) {
    __shared__ float sv[4096];
    __shared__ float partial[256];
    __shared__ float yv[64];
    __shared__ float gv[64];
    __shared__ float stats[2];

    const int b   = blockIdx.x / M;
    const int t   = blockIdx.x % M;
    const int tid = threadIdx.x;
    const float* xb = x + (size_t)b * N_ * 64;

    int cat = 0, E = 0, py = 0, px = 0, cmul = 1, cadd = 0, src = 0;
    const float* W  = wA;
    const float* bs = bA;
    if (branch == 0) {
        if (t < 64)       { cat = 1; E = 4096; py = t >> 3; px = t & 7; W = wA; bs = bA; }
        else if (t < 192) { int u = t - 64;  cat = 2; E = 2048; int s = u & 63; py = s >> 3; px = s & 7; cmul = 2; cadd = u >> 6; W = wB; bs = bB; }
        else if (t < 448) { int u = t - 192; cat = 3; E = 1024; int s = u & 63; py = s >> 3; px = s & 7; cmul = 4; cadd = u >> 6; W = wC; bs = bC; }
        else              { src = 7168 + (t - 448); }
    } else {
        if (t < 256)       { cat = 1; E = 1024; py = t >> 4; px = t & 15; W = wA; bs = bA; }
        else if (t < 768)  { int u = t - 256; cat = 2; E = 512; int s = u & 255; py = s >> 4; px = s & 15; cmul = 2; cadd = u >> 8; W = wB; bs = bB; }
        else if (t < 1792) { int u = t - 768; cat = 3; E = 256; int s = u & 255; py = s >> 4; px = s & 15; cmul = 4; cadd = u >> 8; W = wC; bs = bC; }
        else               { src = 7168 + (t - 1792); }
    }

    // Phase 1: gather patch vector (or passthrough slab token).
    if (cat == 0) {
        if (tid < 64) yv[tid] = xb[(size_t)src * 64 + tid];
    } else {
        for (int e = tid; e < E; e += 256) {
            int token, feat;
            if (branch == 0) {
                if (cat == 1)      { int ci = e >> 6, rem = e & 63, ky = rem >> 3, kx = rem & 7;
                                     token = (py * 8 + ky) * 64 + px * 8 + kx; feat = ci; }
                else if (cat == 2) { int ci = e >> 4, rem = e & 15, ky = rem >> 2, kx = rem & 3;
                                     token = 4096 + (py * 4 + ky) * 64 + 2 * (px * 4 + kx) + (ci >> 6); feat = ci & 63; }
                else               { int ci = e >> 2, rem = e & 3, ky = rem >> 1, kx = rem & 1;
                                     token = 6144 + (py * 2 + ky) * 64 + 4 * (px * 2 + kx) + (ci >> 6); feat = ci & 63; }
            } else {
                if (cat == 1)      { int ci = e >> 4, rem = e & 15, ky = rem >> 2, kx = rem & 3;
                                     token = (py * 4 + ky) * 64 + px * 4 + kx; feat = ci; }
                else if (cat == 2) { int ci = e >> 2, rem = e & 3, ky = rem >> 1, kx = rem & 1;
                                     token = 4096 + (py * 2 + ky) * 64 + 2 * (px * 2 + kx) + (ci >> 6); feat = ci & 63; }
                else               { int ci = e;
                                     token = 6144 + py * 64 + 4 * px + (ci >> 6); feat = ci & 63; }
            }
            sv[e] = xb[(size_t)token * 64 + feat];
        }
    }
    __syncthreads();

    // Phase 2: conv as 64 x E matmul (4 partial lanes per output feature).
    if (cat != 0) {
        const int f = tid & 63, part = tid >> 6;
        const int c = cmul * f + cadd;
        const float* wr = W + (size_t)c * E;
        float acc = 0.f;
        for (int e = part; e < E; e += 4) acc = fmaf(wr[e], sv[e], acc);
        partial[tid] = acc;
    }
    __syncthreads();
    if (cat != 0 && tid < 64) {
        yv[tid] = bs[cmul * tid + cadd] + partial[tid] + partial[tid + 64] +
                  partial[tid + 128] + partial[tid + 192];
    }
    __syncthreads();

    // Phase 3: LayerNorm + exact GELU.
    if (tid == 0) {
        float mu = 0.f;
        for (int c = 0; c < 64; ++c) mu += yv[c];
        mu *= (1.f / 64.f);
        float var = 0.f;
        for (int c = 0; c < 64; ++c) { float dlt = yv[c] - mu; var = fmaf(dlt, dlt, var); }
        var *= (1.f / 64.f);
        stats[0] = mu;
        stats[1] = rsqrtf(var + 1e-5f);
    }
    __syncthreads();
    if (tid < 64) {
        float zn = (yv[tid] - stats[0]) * stats[1] * lnw[tid] + lnb[tid];
        gv[tid] = 0.5f * zn * (1.f + erff(zn * 0.70710678118654752f));
    }
    __syncthreads();

    // Phase 4: kv projection -> K row-major / V transposed.
    if (tid < 64) {
        float acc = kvb[tid];
#pragma unroll 8
        for (int c = 0; c < 64; ++c) acc = fmaf(gv[c], kvw[c * 64 + tid], acc);
        if (tid < 32) K[((size_t)b * M + t) * 32 + tid] = (_Float16)acc;
        else          VT[((size_t)(b * 32 + (tid - 32))) * M + t] = (_Float16)acc;
    }
}

// ---------------------------------------------------------------------------
// Kernel 3: WMMA flash attention.  256 threads = 8 waves; each wave owns a
// 16-row Q tile and streams K/V in 64-key chunks: 4 WMMAs for the 16x64 score
// tile, one online-softmax pass (DPP16 row butterflies amortized over 64
// keys), P -> LDS -> two A-layout operands, 4 WMMAs accumulating O in f32.
// ---------------------------------------------------------------------------
__global__ void flash_kernel(const _Float16* __restrict__ Q,   // [2][B*N][32]
                             const _Float16* __restrict__ K1,
                             const _Float16* __restrict__ V1T,
                             const _Float16* __restrict__ K2,
                             const _Float16* __restrict__ V2T,
                             float* __restrict__ O) {          // [B*N][64]
    __shared__ u16 ldsP[8][16 * 64];

    const int branch = blockIdx.y;
    const int tile   = blockIdx.x * 8 + (threadIdx.x >> 5);  // over (B*N)/16
    const int lane   = threadIdx.x & 31;
    const int half   = lane >> 4;
    const int n      = lane & 15;
    const int rows   = tile * 16;                            // flat over B*N
    const int b      = rows / N_;
    const int M      = branch ? M2_ : M1_;
    const _Float16* Kp = (branch ? K2 : K1) + (size_t)b * M * 32;
    const _Float16* Vt = (branch ? V2T : V1T) + (size_t)b * 32 * M;
    const _Float16* Qp = Q + (size_t)branch * (B_ * N_) * 32;

    // A operand: 16x32 Q tile.  Lane row = lane%16; VGPR v holds the f16 pair
    // at feature k0 = 2v + 16*half  -> contiguous b128 loads.
    H16 aq;
    {
        const _Float16* qrow = Qp + (size_t)(rows + n) * 32;
#pragma unroll
        for (int v = 0; v < 8; ++v) {
            const int k0 = 2 * v + 16 * half;
            aq.u[v] = *(const u32*)(qrow + k0);
        }
    }

    F8 acc0, acc1;
    float mrow[8], lrow[8];
#pragma unroll
    for (int r = 0; r < 8; ++r) {
        acc0.f[r] = 0.f; acc1.f[r] = 0.f;
        mrow[r] = -INFINITY; lrow[r] = 0.f;
    }
    v8f zero = {};
    u16* pbuf = ldsP[threadIdx.x >> 5];

    for (int kb = 0; kb < M; kb += 64) {
        // B operands for K: column n = key token; 16 consecutive f16 per lane.
        H16 bk0, bk1, bk2, bk3;
        const _Float16* kr0 = Kp + (size_t)(kb + n) * 32 + 16 * half;
        const _Float16* kr1 = kr0 + 16 * 32;
        const _Float16* kr2 = kr0 + 32 * 32;
        const _Float16* kr3 = kr0 + 48 * 32;
#pragma unroll
        for (int v = 0; v < 8; ++v) {
            bk0.u[v] = *(const u32*)(kr0 + 2 * v);
            bk1.u[v] = *(const u32*)(kr1 + 2 * v);
            bk2.u[v] = *(const u32*)(kr2 + 2 * v);
            bk3.u[v] = *(const u32*)(kr3 + 2 * v);
        }
        // B operands for V issued early so the b128 loads overlap the softmax
        // VALU/TRANS work.  V stored transposed -> consecutive tokens are
        // contiguous dwords.  bv0/bv1: tokens kb..kb+31, bv2/bv3: kb+32..+63.
        H16 bv0, bv1, bv2, bv3;
        const _Float16* vr0 = Vt + (size_t)n * M + kb + 16 * half;
        const _Float16* vr1 = Vt + (size_t)(16 + n) * M + kb + 16 * half;
#pragma unroll
        for (int v = 0; v < 8; ++v) {
            bv0.u[v] = *(const u32*)(vr0 + 2 * v);
            bv1.u[v] = *(const u32*)(vr1 + 2 * v);
            bv2.u[v] = *(const u32*)(vr0 + 32 + 2 * v);
            bv3.u[v] = *(const u32*)(vr1 + 32 + 2 * v);
        }
        if (kb + 64 < M) {  // global_prefetch_b8 next K/V chunks
            __builtin_prefetch(kr0 + 64 * 32, 0, 1);
            __builtin_prefetch(kr2 + 64 * 32, 0, 1);
            __builtin_prefetch(vr0 + 64, 0, 1);
            __builtin_prefetch(vr1 + 64, 0, 1);
        }

        F8 s0, s1, s2, s3;
        s0.v = __builtin_amdgcn_wmma_f32_16x16x32_f16(false, aq.v, false, bk0.v,
                                                      (short)0, zero, false, false);
        s1.v = __builtin_amdgcn_wmma_f32_16x16x32_f16(false, aq.v, false, bk1.v,
                                                      (short)0, zero, false, false);
        s2.v = __builtin_amdgcn_wmma_f32_16x16x32_f16(false, aq.v, false, bk2.v,
                                                      (short)0, zero, false, false);
        s3.v = __builtin_amdgcn_wmma_f32_16x16x32_f16(false, aq.v, false, bk3.v,
                                                      (short)0, zero, false, false);

        // Online softmax over the 16x64 tile.  C layout: element
        // (row r + 8*half, col n); one DPP butterfly pair per row per chunk.
#pragma unroll
        for (int r = 0; r < 8; ++r) {
            const float cm   = row_max16(fmaxf(fmaxf(s0.f[r], s1.f[r]),
                                               fmaxf(s2.f[r], s3.f[r])));
            const float mnew = fmaxf(mrow[r], cm);
            const float corr = __expf(mrow[r] - mnew);
            mrow[r] = mnew;
            acc0.f[r] *= corr;
            acc1.f[r] *= corr;
            const float p0 = __expf(s0.f[r] - mnew);
            const float p1 = __expf(s1.f[r] - mnew);
            const float p2 = __expf(s2.f[r] - mnew);
            const float p3 = __expf(s3.f[r] - mnew);
            lrow[r] = lrow[r] * corr + row_sum16((p0 + p1) + (p2 + p3));
            const int m = r + 8 * half;
            pbuf[m * 64 + n]      = f32_to_h16(p0);
            pbuf[m * 64 + 16 + n] = f32_to_h16(p1);
            pbuf[m * 64 + 32 + n] = f32_to_h16(p2);
            pbuf[m * 64 + 48 + n] = f32_to_h16(p3);
        }
        asm volatile("s_wait_dscnt 0" ::: "memory");

        // P back in A layout: contiguous 8-f16 runs -> wide DS loads
        // (memcpy is alias-safe against the u16 stores above).
        // ap0 = P[:,0:32], ap1 = P[:,32:64].
        H16 ap0, ap1;
        __builtin_memcpy(&ap0.u[0], &pbuf[n * 64 + 16 * half], 16);
        __builtin_memcpy(&ap0.u[4], &pbuf[n * 64 + 8 + 16 * half], 16);
        __builtin_memcpy(&ap1.u[0], &pbuf[n * 64 + 32 + 16 * half], 16);
        __builtin_memcpy(&ap1.u[4], &pbuf[n * 64 + 40 + 16 * half], 16);

        acc0.v = __builtin_amdgcn_wmma_f32_16x16x32_f16(false, ap0.v, false, bv0.v,
                                                        (short)0, acc0.v, false, false);
        acc1.v = __builtin_amdgcn_wmma_f32_16x16x32_f16(false, ap0.v, false, bv1.v,
                                                        (short)0, acc1.v, false, false);
        acc0.v = __builtin_amdgcn_wmma_f32_16x16x32_f16(false, ap1.v, false, bv2.v,
                                                        (short)0, acc0.v, false, false);
        acc1.v = __builtin_amdgcn_wmma_f32_16x16x32_f16(false, ap1.v, false, bv3.v,
                                                        (short)0, acc1.v, false, false);
    }

    // Epilogue: O[row] = acc / l.
#pragma unroll
    for (int r = 0; r < 8; ++r) {
        const float inv = 1.f / lrow[r];
        const int m = r + 8 * half;
        const size_t base = ((size_t)(rows + m)) * 64 + branch * 32;
        O[base + n]      = acc0.f[r] * inv;
        O[base + 16 + n] = acc1.f[r] * inv;
    }
}

// ---------------------------------------------------------------------------
// Kernel 4: output projection  out = O @ proj_w + proj_b  (f32).
// ---------------------------------------------------------------------------
__global__ void proj_kernel(const float* __restrict__ Oin,
                            const float* __restrict__ pw,
                            const float* __restrict__ pb,
                            float* __restrict__ out) {
    __shared__ float xs[4][64];
    const int slot  = threadIdx.x >> 6;
    const int j     = threadIdx.x & 63;
    const int token = blockIdx.x * 4 + slot;
    xs[slot][j] = Oin[(size_t)token * 64 + j];
    __syncthreads();
    const float* xr = xs[slot];
    float acc = pb[j];
#pragma unroll 8
    for (int c = 0; c < 64; ++c) acc = fmaf(xr[c], pw[c * 64 + j], acc);
    out[(size_t)token * 64 + j] = acc;
}

// ---------------------------------------------------------------------------
extern "C" void kernel_launch(void* const* d_in, const int* in_sizes, int n_in,
                              void* d_out, int out_size, void* d_ws, size_t ws_size,
                              hipStream_t stream) {
    (void)in_sizes; (void)n_in; (void)out_size; (void)ws_size;
    const float* x      = (const float*)d_in[0];
    const float* q_w    = (const float*)d_in[1];
    const float* q_b    = (const float*)d_in[2];
    const float* kv1_w  = (const float*)d_in[3];
    const float* kv1_b  = (const float*)d_in[4];
    const float* kv2_w  = (const float*)d_in[5];
    const float* kv2_b  = (const float*)d_in[6];
    const float* proj_w = (const float*)d_in[7];
    const float* proj_b = (const float*)d_in[8];
    const float* sr1_w  = (const float*)d_in[9];
    const float* sr1_b  = (const float*)d_in[10];
    const float* sr2_w  = (const float*)d_in[11];
    const float* sr2_b  = (const float*)d_in[12];
    const float* sr3_w  = (const float*)d_in[13];
    const float* sr3_b  = (const float*)d_in[14];
    const float* sr4_w  = (const float*)d_in[15];
    const float* sr4_b  = (const float*)d_in[16];
    const float* sr5_w  = (const float*)d_in[17];
    const float* sr5_b  = (const float*)d_in[18];
    const float* sr6_w  = (const float*)d_in[19];
    const float* sr6_b  = (const float*)d_in[20];
    const float* ln1_w  = (const float*)d_in[21];
    const float* ln1_b  = (const float*)d_in[22];
    const float* ln2_w  = (const float*)d_in[23];
    const float* ln2_b  = (const float*)d_in[24];

    char* ws = (char*)d_ws;
    size_t off = 0;
    auto wsAlloc = [&](size_t bytes) -> char* {
        char* p = ws + off;
        off += (bytes + 255) & ~(size_t)255;
        return p;
    };
    _Float16* Qh  = (_Float16*)wsAlloc((size_t)2 * B_ * N_ * 32 * sizeof(_Float16));
    _Float16* K1  = (_Float16*)wsAlloc((size_t)B_ * M1_ * 32 * sizeof(_Float16));
    _Float16* V1T = (_Float16*)wsAlloc((size_t)B_ * 32 * M1_ * sizeof(_Float16));
    _Float16* K2  = (_Float16*)wsAlloc((size_t)B_ * M2_ * 32 * sizeof(_Float16));
    _Float16* V2T = (_Float16*)wsAlloc((size_t)B_ * 32 * M2_ * sizeof(_Float16));
    float*    Obf = (float*)wsAlloc((size_t)B_ * N_ * 64 * sizeof(float));

    qproj_kernel<<<B_ * N_ / 4, 256, 0, stream>>>(x, q_w, q_b, Qh);
    build_kv_kernel<<<B_ * M1_, 256, 0, stream>>>(x, sr1_w, sr1_b, sr3_w, sr3_b,
                                                  sr5_w, sr5_b, ln1_w, ln1_b,
                                                  kv1_w, kv1_b, K1, V1T, M1_, 0);
    build_kv_kernel<<<B_ * M2_, 256, 0, stream>>>(x, sr2_w, sr2_b, sr4_w, sr4_b,
                                                  sr6_w, sr6_b, ln2_w, ln2_b,
                                                  kv2_w, kv2_b, K2, V2T, M2_, 1);
    flash_kernel<<<dim3(B_ * N_ / 128, 2), 256, 0, stream>>>(Qh, K1, V1T, K2, V2T, Obf);
    proj_kernel<<<B_ * N_ / 4, 256, 0, stream>>>(Obf, proj_w, proj_b, (float*)d_out);
}
